// Net_26938034880515
// MI455X (gfx1250) — compile-verified, matrix operand order checked by reading, not run on hardware
//
#include <hip/hip_runtime.h>
#include <hip/hip_bf16.h>
#include <math.h>

#define N_NODES   65536
#define N_EDGES   1048576
#define N_GRAPHS  1024
#define DIM       70
#define K12       840          // 12*DIM
#define KPAD      864          // 27 * 32
#define NPAD      80           // 5 * 16
#define NUM_LAYERS 4
#define AVG_LOG   2.8332133441f
#define BN_EPS    1e-5f

typedef _Float16 v16h __attribute__((ext_vector_type(16)));
typedef _Float16 v8h  __attribute__((ext_vector_type(8)));
typedef float    v8f  __attribute__((ext_vector_type(8)));

// ---- order-preserving float <-> uint encoding for atomic min/max ----
__device__ __forceinline__ unsigned f2ord(float f) {
    unsigned u = __float_as_uint(f);
    return (u & 0x80000000u) ? ~u : (u | 0x80000000u);
}
__device__ __forceinline__ float ord2f(unsigned u) {
    return (u & 0x80000000u) ? __uint_as_float(u & 0x7FFFFFFFu)
                             : __uint_as_float(~u);
}

// ================= atom encoder: h[n][d] = sum_f atom_emb[f][x[n][f]][d] ====
__global__ void k_atom(const int* __restrict__ x, const float* __restrict__ emb,
                       float* __restrict__ h) {
    int tid  = blockIdx.x * blockDim.x + threadIdx.x;
    int nd   = tid >> 5;
    int lane = tid & 31;
    float a0 = 0.f, a1 = 0.f, a2 = 0.f;
#pragma unroll
    for (int f = 0; f < 9; ++f) {
        int idx = x[nd * 9 + f];
        const float* e = emb + (size_t)(f * 119 + idx) * DIM;
        a0 += e[lane];
        a1 += e[lane + 32];
        if (lane < 6) a2 += e[lane + 64];
    }
    float* row = h + (size_t)nd * DIM;
    row[lane] = a0; row[lane + 32] = a1;
    if (lane < 6) row[lane + 64] = a2;
}

// ================= bond encoder =============================================
__global__ void k_bond(const int* __restrict__ ea, const float* __restrict__ emb,
                       float* __restrict__ ee) {
    int tid  = blockIdx.x * blockDim.x + threadIdx.x;
    int e    = tid >> 5;
    int lane = tid & 31;
    float a0 = 0.f, a1 = 0.f, a2 = 0.f;
#pragma unroll
    for (int f = 0; f < 3; ++f) {
        int idx = ea[e * 3 + f];
        const float* p = emb + (size_t)(f * 5 + idx) * DIM;
        a0 += p[lane];
        a1 += p[lane + 32];
        if (lane < 6) a2 += p[lane + 64];
    }
    float* row = ee + (size_t)e * DIM;
    row[lane] = a0; row[lane + 32] = a1;
    if (lane < 6) row[lane + 64] = a2;
}

// ================= init aggregation buffers =================================
__global__ void k_init_agg(float* __restrict__ s, float* __restrict__ q,
                           unsigned* __restrict__ mnb, unsigned* __restrict__ mxb,
                           float* __restrict__ deg) {
    int i = blockIdx.x * blockDim.x + threadIdx.x;
    if (i < N_NODES * DIM) {
        s[i] = 0.f; q[i] = 0.f; mnb[i] = 0xFFFFFFFFu; mxb[i] = 0u;
    }
    if (i < N_NODES) deg[i] = 0.f;
}

// ================= edge scatter: one wave per edge ==========================
__global__ void k_edge(const int* __restrict__ ei, const float* __restrict__ h,
                       const float* __restrict__ ee,
                       float* __restrict__ s, float* __restrict__ q,
                       unsigned* __restrict__ mnb, unsigned* __restrict__ mxb,
                       float* __restrict__ deg) {
    int tid  = blockIdx.x * blockDim.x + threadIdx.x;
    int e    = tid >> 5;
    int lane = tid & 31;
    int src = ei[e];
    int dst = ei[N_EDGES + e];
    const float* hs = h  + (size_t)src * DIM;
    const float* em = ee + (size_t)e   * DIM;
    size_t ob = (size_t)dst * DIM;
#pragma unroll
    for (int c = 0; c < 3; ++c) {
        int d = lane + c * 32;
        if (d < DIM) {
            float m = hs[d] + em[d];
            atomicAdd(&s[ob + d], m);
            atomicAdd(&q[ob + d], m * m);
            unsigned o = f2ord(m);
            atomicMin(&mnb[ob + d], o);
            atomicMax(&mxb[ob + d], o);
        }
    }
    if (lane == 0) atomicAdd(&deg[dst], 1.0f);
}

// ================= finalize: build f16 A row [KPAD] per node ================
__device__ __forceinline__ void fin_one(int d, size_t ob, float inv, float amp,
                                        float att, bool has,
                                        const float* __restrict__ s,
                                        const float* __restrict__ q,
                                        const unsigned* __restrict__ mnb,
                                        const unsigned* __restrict__ mxb,
                                        _Float16* __restrict__ row) {
    float mean = s[ob + d] * inv;
    float sqm  = q[ob + d] * inv;
    float sd   = sqrtf(fmaxf(sqm - mean * mean, 0.f) + 1e-5f);
    float mn   = has ? ord2f(mnb[ob + d]) : 0.f;
    float mx   = has ? ord2f(mxb[ob + d]) : 0.f;
    row[d]             = (_Float16)mean;
    row[d + 70]        = (_Float16)mn;
    row[d + 140]       = (_Float16)mx;
    row[d + 210]       = (_Float16)sd;
    row[d + 280]       = (_Float16)(mean * amp);
    row[d + 350]       = (_Float16)(mn * amp);
    row[d + 420]       = (_Float16)(mx * amp);
    row[d + 490]       = (_Float16)(sd * amp);
    row[d + 560]       = (_Float16)(mean * att);
    row[d + 630]       = (_Float16)(mn * att);
    row[d + 700]       = (_Float16)(mx * att);
    row[d + 770]       = (_Float16)(sd * att);
}

__global__ void k_finalize(const float* __restrict__ s, const float* __restrict__ q,
                           const unsigned* __restrict__ mnb,
                           const unsigned* __restrict__ mxb,
                           const float* __restrict__ deg,
                           _Float16* __restrict__ A) {
    int tid  = blockIdx.x * blockDim.x + threadIdx.x;
    int nd   = tid >> 5;
    int lane = tid & 31;
    float dg  = deg[nd];
    float d1  = fmaxf(dg, 1.f);
    float inv = 1.f / d1;
    float lg  = logf(d1 + 1.f);
    float amp = lg / AVG_LOG;
    float att = AVG_LOG / lg;
    bool  has = dg > 0.f;
    size_t ob = (size_t)nd * DIM;
    _Float16* row = A + (size_t)nd * KPAD;
    fin_one(lane,      ob, inv, amp, att, has, s, q, mnb, mxb, row);
    fin_one(lane + 32, ob, inv, amp, att, has, s, q, mnb, mxb, row);
    if (lane < 6) fin_one(lane + 64, ob, inv, amp, att, has, s, q, mnb, mxb, row);
    if (lane < 24) row[K12 + lane] = (_Float16)0.f;   // K padding
}

// ================= pack W^T into f16 [NPAD][KPAD] ===========================
__global__ void k_packW(const float* __restrict__ W, _Float16* __restrict__ Wt) {
    int i = blockIdx.x * blockDim.x + threadIdx.x;
    if (i >= NPAD * KPAD) return;
    int n = i / KPAD, k = i - n * KPAD;
    float v = (n < DIM && k < K12) ? W[(size_t)k * DIM + n] : 0.f;
    Wt[i] = (_Float16)v;
}

// ================= WMMA GEMM: y[N,70] = A[N,864] @ Wt^T + bias ==============
// one wave -> 16 rows x 80 cols (5 n-tiles), K-loop of 27 wmma steps/tile
__global__ void k_gemm(const _Float16* __restrict__ A,
                       const _Float16* __restrict__ Wt,
                       const float* __restrict__ bias,
                       float* __restrict__ y) {
    int tid  = blockIdx.x * blockDim.x + threadIdx.x;
    int wave = tid >> 5;
    int lane = tid & 31;
    int row0 = wave * 16;
    int half = lane >> 4;          // 0: K 0-7/16-23, 1: K 8-15/24-31
    int mrow = lane & 15;
    int koff = half * 8;
    const _Float16* arow = A + (size_t)(row0 + mrow) * KPAD + koff;

    v8f acc[5];
#pragma unroll
    for (int nt = 0; nt < 5; ++nt)
#pragma unroll
        for (int r = 0; r < 8; ++r) acc[nt][r] = 0.f;

    for (int kt = 0; kt < KPAD / 32; ++kt) {
        v8h alo = *(const v8h*)(arow + kt * 32);
        v8h ahi = *(const v8h*)(arow + kt * 32 + 16);
        v16h a;
#pragma unroll
        for (int i = 0; i < 8; ++i) { a[i] = alo[i]; a[i + 8] = ahi[i]; }
#pragma unroll
        for (int nt = 0; nt < 5; ++nt) {
            const _Float16* brow =
                Wt + (size_t)(nt * 16 + mrow) * KPAD + koff + kt * 32;
            v8h blo = *(const v8h*)brow;
            v8h bhi = *(const v8h*)(brow + 16);
            v16h b;
#pragma unroll
            for (int i = 0; i < 8; ++i) { b[i] = blo[i]; b[i + 8] = bhi[i]; }
            acc[nt] = __builtin_amdgcn_wmma_f32_16x16x32_f16(
                false, a, false, b, (short)0, acc[nt], false, false);
        }
    }
    // epilogue: D layout -> lane n = lane&15, M = half*8 + r
    int m0 = half * 8;
#pragma unroll
    for (int nt = 0; nt < 5; ++nt) {
        int col = nt * 16 + (lane & 15);
        if (col < DIM) {
            float bv = bias[col];
#pragma unroll
            for (int r = 0; r < 8; ++r)
                y[(size_t)(row0 + m0 + r) * DIM + col] = acc[nt][r] + bv;
        }
    }
}

// ================= BN batch statistics (per column) =========================
__global__ void k_bnstats(const float* __restrict__ y, float* __restrict__ mu,
                          float* __restrict__ var) {
    __shared__ float ss[256], sq[256];
    int col = blockIdx.x, t = threadIdx.x;
    float s = 0.f, q = 0.f;
    for (int r = t; r < N_NODES; r += 256) {
        float v = y[(size_t)r * DIM + col];
        s += v; q += v * v;
    }
    ss[t] = s; sq[t] = q; __syncthreads();
    for (int off = 128; off; off >>= 1) {
        if (t < off) { ss[t] += ss[t + off]; sq[t] += sq[t + off]; }
        __syncthreads();
    }
    if (t == 0) {
        float m = ss[0] / (float)N_NODES;
        mu[col]  = m;
        var[col] = sq[0] / (float)N_NODES - m * m;
    }
}

// ================= BN apply + relu + residual into h ========================
__global__ void k_bnapply(const float* __restrict__ y, const float* __restrict__ mu,
                          const float* __restrict__ var,
                          const float* __restrict__ gamma,
                          const float* __restrict__ beta,
                          float* __restrict__ h) {
    int i = blockIdx.x * blockDim.x + threadIdx.x;
    if (i >= N_NODES * DIM) return;
    int col = i % DIM;
    float v = (y[i] - mu[col]) * rsqrtf(var[col] + BN_EPS) * gamma[col] + beta[col];
    h[i] = fmaxf(v, 0.f) + h[i];
}

// ================= pooling ==================================================
__global__ void k_zero_pool(float* __restrict__ pooled, float* __restrict__ cnt) {
    int i = blockIdx.x * blockDim.x + threadIdx.x;
    if (i < N_GRAPHS * DIM) pooled[i] = 0.f;
    if (i < N_GRAPHS) cnt[i] = 0.f;
}

__global__ void k_pool(const float* __restrict__ h, const int* __restrict__ batch,
                       float* __restrict__ pooled, float* __restrict__ cnt) {
    int tid  = blockIdx.x * blockDim.x + threadIdx.x;
    int nd   = tid >> 5;
    int lane = tid & 31;
    int g = batch[nd];
    const float* row = h + (size_t)nd * DIM;
    size_t ob = (size_t)g * DIM;
#pragma unroll
    for (int c = 0; c < 3; ++c) {
        int d = lane + c * 32;
        if (d < DIM) atomicAdd(&pooled[ob + d], row[d]);
    }
    if (lane == 0) atomicAdd(&cnt[g], 1.0f);
}

// ================= final tiny MLP: one thread per graph =====================
__global__ void k_mlp(const float* __restrict__ pooled, const float* __restrict__ cnt,
                      const float* __restrict__ W1, const float* __restrict__ b1,
                      const float* __restrict__ W2, const float* __restrict__ b2,
                      const float* __restrict__ W3, const float* __restrict__ b3,
                      float* __restrict__ out) {
    int g = blockIdx.x * blockDim.x + threadIdx.x;
    if (g >= N_GRAPHS) return;
    float c = fmaxf(cnt[g], 1.f);
    float p[DIM];
    for (int d = 0; d < DIM; ++d) p[d] = pooled[(size_t)g * DIM + d] / c;
    float z1[35];
    for (int j = 0; j < 35; ++j) {
        float z = b1[j];
        for (int d = 0; d < DIM; ++d) z += p[d] * W1[d * 35 + j];
        z1[j] = fmaxf(z, 0.f);
    }
    float z2[17];
    for (int k = 0; k < 17; ++k) {
        float z = b2[k];
        for (int j = 0; j < 35; ++j) z += z1[j] * W2[j * 17 + k];
        z2[k] = fmaxf(z, 0.f);
    }
    float z = b3[0];
    for (int k = 0; k < 17; ++k) z += z2[k] * W3[k];
    out[g] = z;
}

// ============================================================================
extern "C" void kernel_launch(void* const* d_in, const int* in_sizes, int n_in,
                              void* d_out, int out_size, void* d_ws, size_t ws_size,
                              hipStream_t stream) {
    const int*   x         = (const int*)  d_in[0];
    const int*   ei        = (const int*)  d_in[1];
    const int*   ea        = (const int*)  d_in[2];
    const int*   batch     = (const int*)  d_in[3];
    const float* atom_emb  = (const float*)d_in[4];
    const float* bond_emb  = (const float*)d_in[5];
    const float* post_W    = (const float*)d_in[6];
    const float* post_b    = (const float*)d_in[7];
    const float* bn_gamma  = (const float*)d_in[8];
    const float* bn_beta   = (const float*)d_in[9];
    const float* mlp_W1    = (const float*)d_in[10];
    const float* mlp_b1    = (const float*)d_in[11];
    const float* mlp_W2    = (const float*)d_in[12];
    const float* mlp_b2    = (const float*)d_in[13];
    const float* mlp_W3    = (const float*)d_in[14];
    const float* mlp_b3    = (const float*)d_in[15];
    float* out = (float*)d_out;

    // ---- carve workspace ----
    char* base = (char*)d_ws;
    size_t off = 0;
    auto carve = [&](size_t bytes) -> char* {
        char* p = base + off;
        off = (off + bytes + 255) & ~(size_t)255;
        return p;
    };
    float*     h      = (float*)    carve((size_t)N_NODES * DIM * 4);
    float*     ee     = (float*)    carve((size_t)N_EDGES * DIM * 4);
    float*     segs   = (float*)    carve((size_t)N_NODES * DIM * 4);
    float*     segq   = (float*)    carve((size_t)N_NODES * DIM * 4);
    unsigned*  mnb    = (unsigned*) carve((size_t)N_NODES * DIM * 4);
    unsigned*  mxb    = (unsigned*) carve((size_t)N_NODES * DIM * 4);
    float*     deg    = (float*)    carve((size_t)N_NODES * 4);
    _Float16*  Amat   = (_Float16*) carve((size_t)N_NODES * KPAD * 2);
    _Float16*  Wt     = (_Float16*) carve((size_t)NPAD * KPAD * 2);
    float*     y      = (float*)    carve((size_t)N_NODES * DIM * 4);
    float*     mu     = (float*)    carve(DIM * 4);
    float*     var    = (float*)    carve(DIM * 4);
    float*     pooled = (float*)    carve((size_t)N_GRAPHS * DIM * 4);
    float*     cnt    = (float*)    carve((size_t)N_GRAPHS * 4);
    (void)ws_size; (void)n_in; (void)in_sizes; (void)out_size;

    const int T = 256;
    int blk_nodeW = (N_NODES * 32) / T;   // wave-per-node kernels
    int blk_edgeW = (N_EDGES * 32) / T;   // wave-per-edge kernels
    int blk_nd    = (N_NODES * DIM + T - 1) / T;

    // encoders
    k_atom<<<blk_nodeW, T, 0, stream>>>(x, atom_emb, h);
    k_bond<<<blk_edgeW, T, 0, stream>>>(ea, bond_emb, ee);

    for (int l = 0; l < NUM_LAYERS; ++l) {
        k_init_agg<<<blk_nd, T, 0, stream>>>(segs, segq, mnb, mxb, deg);
        k_edge<<<blk_edgeW, T, 0, stream>>>(ei, h, ee, segs, segq, mnb, mxb, deg);
        k_finalize<<<blk_nodeW, T, 0, stream>>>(segs, segq, mnb, mxb, deg, Amat);
        k_packW<<<(NPAD * KPAD + T - 1) / T, T, 0, stream>>>(
            post_W + (size_t)l * K12 * DIM, Wt);
        k_gemm<<<(N_NODES / 16) * 32 / T, T, 0, stream>>>(
            Amat, Wt, post_b + (size_t)l * DIM, y);
        k_bnstats<<<DIM, T, 0, stream>>>(y, mu, var);
        k_bnapply<<<blk_nd, T, 0, stream>>>(
            y, mu, var, bn_gamma + (size_t)l * DIM, bn_beta + (size_t)l * DIM, h);
    }

    k_zero_pool<<<(N_GRAPHS * DIM + T - 1) / T, T, 0, stream>>>(pooled, cnt);
    k_pool<<<blk_nodeW, T, 0, stream>>>(h, batch, pooled, cnt);
    k_mlp<<<(N_GRAPHS + T - 1) / T, T, 0, stream>>>(
        pooled, cnt, mlp_W1, mlp_b1, mlp_W2, mlp_b2, mlp_W3, mlp_b3, out);
}
